// LinkPredictor_41308995452995
// MI455X (gfx1250) — matrix operand order repaired and checked by reading, not compiled, not run on hardware
//
#include <hip/hip_runtime.h>

typedef _Float16 f16;
typedef __attribute__((ext_vector_type(16))) _Float16 v16h;
typedef __attribute__((ext_vector_type(8)))  _Float16 v8h;
typedef __attribute__((ext_vector_type(4)))  _Float16 v4h;
typedef __attribute__((ext_vector_type(8)))  float    v8f;

#define IN_CH    256
#define KDIM     512     // 2*IN_CH
#define HID      256
#define KT       16      // K tiles of 32
#define NT       16      // N tiles of 16
#define M_TILE   128     // edges per block
#define MT       8       // M tiles of 16
#define NWAVES   8
#define A_STRIDE 536     // halfs per LDS A row (1072 B: 16 rows -> 16 distinct bank groups)

// ---------------------------------------------------------------------------
// Pack W1 (f32 [512,256], row-major, k-major) into f16 WMMA B-fragment order.
// Fragment (kt, nt): lane l holds 16 consecutive K values of column
// n = nt*16 + (l&15), starting at K = kt*32 + (l>=16 ? 16 : 0).
// Storage: frag-major, lane-major, 16 halfs (32 B) per lane.
// ---------------------------------------------------------------------------
__global__ __launch_bounds__(256) void pack_w1_kernel(const float* __restrict__ W1,
                                                      f16* __restrict__ W1p) {
  int tid  = blockIdx.x * 256 + threadIdx.x;    // 8192 = KT*NT*32
  int lane = tid & 31;
  int nt   = (tid >> 5) & 15;
  int kt   = tid >> 9;
  int n    = nt * 16 + (lane & 15);
  int kb   = kt * 32 + ((lane >> 4) << 4);
  union { v16h v; f16 e[16]; } frag;
  #pragma unroll
  for (int j = 0; j < 16; ++j)
    frag.e[j] = (f16)W1[(size_t)(kb + j) * HID + n];
  *(v16h*)(W1p + (size_t)tid * 16) = frag.v;
}

// ---------------------------------------------------------------------------
// Fused gather + [128,512]x[512,256] GEMM (f16 WMMA, f32 acc) + bias/ReLU +
// second-layer dot with W2 + bias.
// ---------------------------------------------------------------------------
__global__ __launch_bounds__(256) void edge_mlp_kernel(
    const float* __restrict__ emb,
    const long long* __restrict__ eidx,
    const f16* __restrict__ W1p,
    const float* __restrict__ b1,
    const float* __restrict__ W2,
    const float* __restrict__ b2,
    float* __restrict__ out,
    int E)
{
  extern __shared__ __align__(16) unsigned char smem_raw[];
  f16*   Abuf    = (f16*)smem_raw;                                      // M_TILE*A_STRIDE halfs
  float* partial = (float*)(smem_raw + (size_t)M_TILE * A_STRIDE * 2);  // NWAVES*M_TILE f32
  int*   sidx    = (int*)(partial + NWAVES * M_TILE);                   // M_TILE
  int*   didx    = sidx + M_TILE;                                       // M_TILE

  const int tid      = threadIdx.x;
  const int lane     = tid & 31;
  const int wv       = tid >> 5;          // 8 waves
  const int nl       = lane & 15;
  const int tileBase = blockIdx.x * M_TILE;

  // --- stage edge indices ---
  if (tid < M_TILE) {
    int e = tileBase + tid;
    if (e >= E) e = E - 1;                // clamp tail; writes are guarded later
    sidx[tid] = (int)eidx[e];
    didx[tid] = (int)eidx[(size_t)E + e];
  }
  __syncthreads();

  // --- gather: 128 edges x (256 src + 256 dst) f32 -> f16 into LDS A ---
  // 16384 float4 groups, 64 per thread.
  #pragma unroll 8
  for (int g = 0; g < 64; ++g) {
    int gg   = tid + g * 256;
    int edge = gg >> 7;                   // 0..127
    int col  = (gg & 127) << 2;           // 0..508
    const float* p = (col < IN_CH)
        ? emb + (size_t)sidx[edge] * IN_CH + col
        : emb + (size_t)didx[edge] * IN_CH + (col - IN_CH);
    float4 v = *(const float4*)p;
    v4h h; h[0] = (f16)v.x; h[1] = (f16)v.y; h[2] = (f16)v.z; h[3] = (f16)v.w;
    *(v4h*)(Abuf + (size_t)edge * A_STRIDE + col) = h;
  }
  __syncthreads();

  // --- WMMA GEMM: wave wv owns N-tiles {2wv, 2wv+1}, all 8 M-tiles ---
  const int nt0 = wv * 2, nt1 = wv * 2 + 1;
  const int n0  = nt0 * 16 + nl;
  const int n1  = nt1 * 16 + nl;
  const float b1v0 = b1[n0], b1v1 = b1[n1];
  const float w2v0 = W2[n0], w2v1 = W2[n1];

  v8f c[MT][2] = {};
  const int aChunk = (lane >> 4) * 16;    // per-lane K sub-offset within a K-tile

  for (int kt = 0; kt < KT; ++kt) {
    const v16h bf0 = *(const v16h*)(W1p + (((size_t)(kt * NT + nt0) * 32 + lane) << 4));
    const v16h bf1 = *(const v16h*)(W1p + (((size_t)(kt * NT + nt1) * 32 + lane) << 4));
    #pragma unroll
    for (int mt = 0; mt < MT; ++mt) {
      const f16* arow = Abuf + (size_t)(mt * 16 + nl) * A_STRIDE + kt * 32 + aChunk;
      union { v16h v; v8h h[2]; } a;
      a.h[0] = *(const v8h*)(arow);       // K +0..7  (lanes 0-15) / +16..23 (16-31)
      a.h[1] = *(const v8h*)(arow + 8);   // K +8..15 (lanes 0-15) / +24..31 (16-31)
      c[mt][0] = __builtin_amdgcn_wmma_f32_16x16x32_f16(
          false, a.v, false, bf0, (short)0, c[mt][0], false, false);
      c[mt][1] = __builtin_amdgcn_wmma_f32_16x16x32_f16(
          false, a.v, false, bf1, (short)0, c[mt][1], false, false);
    }
  }

  // --- epilogue: relu(h + b1) . W2, reduced over N ---
  // C layout: lane n = lane&15 is the column; VGPR r is row r + (lane>=16 ? 8 : 0).
  #pragma unroll
  for (int mt = 0; mt < MT; ++mt) {
    #pragma unroll
    for (int r = 0; r < 8; ++r) {
      float h0 = c[mt][0][r] + b1v0; h0 = h0 > 0.f ? h0 : 0.f;
      float h1 = c[mt][1][r] + b1v1; h1 = h1 > 0.f ? h1 : 0.f;
      float p  = h0 * w2v0 + h1 * w2v1;
      p += __shfl_xor(p, 1);
      p += __shfl_xor(p, 2);
      p += __shfl_xor(p, 4);
      p += __shfl_xor(p, 8);              // reduce within each 16-lane N group
      if (nl == 0) {
        int m = mt * 16 + r + ((lane >> 4) << 3);
        partial[wv * M_TILE + m] = p;
      }
    }
  }
  __syncthreads();

  if (tid < M_TILE) {
    float s = 0.f;
    #pragma unroll
    for (int w = 0; w < NWAVES; ++w) s += partial[w * M_TILE + tid];
    int e = tileBase + tid;
    if (e < E) out[e] = s + b2[0];
  }
}

// ---------------------------------------------------------------------------
extern "C" void kernel_launch(void* const* d_in, const int* in_sizes, int n_in,
                              void* d_out, int out_size, void* d_ws, size_t ws_size,
                              hipStream_t stream) {
  const float*     emb  = (const float*)d_in[0];
  const long long* eidx = (const long long*)d_in[1];
  const float*     W1   = (const float*)d_in[2];
  const float*     b1   = (const float*)d_in[3];
  const float*     W2   = (const float*)d_in[4];
  const float*     b2   = (const float*)d_in[5];
  float*           out  = (float*)d_out;
  const int E = in_sizes[1] / 2;          // edge_index is [2, E]

  f16* W1p = (f16*)d_ws;                  // 512*256 f16 = 256 KB packed B fragments

  pack_w1_kernel<<<32, 256, 0, stream>>>(W1, W1p);

  const int tiles = (E + M_TILE - 1) / M_TILE;
  const size_t smem = (size_t)M_TILE * A_STRIDE * 2     // A buffer   (137 KB)
                    + (size_t)NWAVES * M_TILE * 4       // partial sums (4 KB)
                    + (size_t)2 * M_TILE * 4;           // src/dst indices (1 KB)
  edge_mlp_kernel<<<tiles, 256, smem, stream>>>(emb, eidx, W1p, b1, W2, b2, out, E);
}
